// LSTMCell_395136991616
// MI455X (gfx1250) — matrix operand order repaired and checked by reading, not compiled, not run on hardware
//
#include <hip/hip_runtime.h>
#include <hip/hip_bf16.h>
#include <math.h>

// ---------------------------------------------------------------------------
// LSTM  B=64, S=256, I=H=1024  ->  hidden_seq (64,256,1024), h(1,64,1024), c(1,64,1024)
//   1) Xg = X @ Wi^T + bi + bh precomputed in parallel (bf16 WMMA, fp32 accum),
//      stored GATE-INTERLEAVED: xg4[row][j][gate] -> one b128 epilogue load.
//   2) ONE persistent kernel runs all 256 recurrence steps:
//        - device-wide barrier between steps (atomic counter + s_sleep spin)
//        - c state lives in registers for the whole sequence
//        - per-block h tile staged in LDS via GLOBAL_LOAD_ASYNC_TO_LDS_B128
//        - 4 gate tiles per wave share one A fragment
//        - next-step Xg prefetched (global_prefetch) before the barrier
// ---------------------------------------------------------------------------

typedef __attribute__((ext_vector_type(8)))  __bf16 v8bf;
typedef __attribute__((ext_vector_type(16))) __bf16 v16bf;
typedef __attribute__((ext_vector_type(8)))  float  v8f;
typedef __attribute__((ext_vector_type(4)))  int    v4i;

// address-space qualified int4 for the async-LDS builtin (signature vV4i*1 V4i*3 Ii Ii)
typedef __attribute__((address_space(1))) v4i as1_v4i;
typedef __attribute__((address_space(3))) v4i as3_v4i;

#define BATCH 64
#define SEQ   256
#define HID   1024
#define G4    4096   // 4*HID
#define NBLK  32     // persistent grid: 32 blocks x 8 waves = 256 tiles
#define NTHR  256

#if __has_builtin(__builtin_amdgcn_global_load_async_to_lds_b128)
#define ASYNC_LDS_COPY 1
#endif

__device__ __forceinline__ unsigned short f32_to_bf16(float f) {
  union { float f; unsigned int u; } v; v.f = f;
  unsigned int lsb = (v.u >> 16) & 1u;
  unsigned int r = v.u + 0x7FFFu + lsb;     // round-to-nearest-even
  return (unsigned short)(r >> 16);
}

// A fragment (16x32 bf16, MxK): lane&15 = row; lanes 0-15 hold K {0..7,16..23},
// lanes 16-31 hold K {8..15,24..31}  (ISA 7.12.2 16-bit A 16x32 layout)
__device__ __forceinline__ v16bf load_a_frag(const unsigned short* __restrict__ base,
                                             int lda, int k0, int lane) {
  int m  = lane & 15;
  int hi = (lane >> 4) & 1;
  const unsigned short* p = base + (size_t)m * lda + k0 + hi * 8;
  v8bf lo  = *reinterpret_cast<const v8bf*>(p);
  v8bf hi8 = *reinterpret_cast<const v8bf*>(p + 16);
  return __builtin_shufflevector(lo, hi8, 0,1,2,3,4,5,6,7,8,9,10,11,12,13,14,15);
}

// B fragment (32x16 bf16, KxN) gathered from row-major W[n][k] (B = W^T):
// lane&15 = column n; lanes 0-15 hold K {0..7,16..23}, lanes 16-31 hold K {8..15,24..31}.
__device__ __forceinline__ v16bf load_b_frag(const unsigned short* __restrict__ wbase,
                                             int ldb, int k0, int lane) {
  int n  = lane & 15;
  int hi = (lane >> 4) & 1;
  const unsigned short* p = wbase + (size_t)n * ldb + k0 + hi * 8;
  v8bf lo  = *reinterpret_cast<const v8bf*>(p);
  v8bf hi8 = *reinterpret_cast<const v8bf*>(p + 16);
  return __builtin_shufflevector(lo, hi8, 0,1,2,3,4,5,6,7,8,9,10,11,12,13,14,15);
}

__device__ __forceinline__ v8f wmma_bf16(v16bf a, v16bf b, v8f c) {
  return __builtin_amdgcn_wmma_f32_16x16x32_bf16(false, a, false, b, (short)0, c,
                                                 false, false);
}

__device__ __forceinline__ float sigmoidf_fast(float x) {
  return 1.0f / (1.0f + __expf(-x));
}

// ---------------------------------------------------------------------------
__global__ void cvt_bf16_kernel(const float* __restrict__ src,
                                unsigned short* __restrict__ dst, int n) {
  int i = blockIdx.x * blockDim.x + threadIdx.x;
  int stride = gridDim.x * blockDim.x;
  for (; i < n; i += stride) dst[i] = f32_to_bf16(src[i]);
}

// init h bf16 working copy + zero the device-wide barrier counter
__global__ void init_state_kernel(const float* __restrict__ h0,
                                  unsigned short* __restrict__ hbf,
                                  unsigned int* __restrict__ bar, int n) {
  int i = blockIdx.x * blockDim.x + threadIdx.x;
  if (i == 0) *bar = 0u;
  if (i < n) hbf[i] = f32_to_bf16(h0[i]);
}

// ---------------------------------------------------------------------------
// xg4[((row*HID)+j)*4 + gate] = sum_k X[row][k]*Wi[g*HID+j][k] + bias; row=b*SEQ+t
// ---------------------------------------------------------------------------
__global__ void input_gemm_kernel(const unsigned short* __restrict__ xbf,
                                  const unsigned short* __restrict__ wibf,
                                  const float* __restrict__ b_i,
                                  const float* __restrict__ b_h,
                                  float* __restrict__ xg4) {
  int lane = threadIdx.x & 31;
  int wave = threadIdx.x >> 5;
  int tile = blockIdx.x * (blockDim.x >> 5) + wave;   // 0 .. 262143
  int bm = tile >> 8;    // 0..1023  row tile
  int bn = tile & 255;   // 0..255   col tile (n = gate*1024 + jj)

  const unsigned short* abase = xbf  + (size_t)(bm * 16) * HID;
  const unsigned short* wbase = wibf + (size_t)(bn * 16) * HID;

  v8f acc = {};
  #pragma unroll 4
  for (int k0 = 0; k0 < HID; k0 += 32) {
    v16bf a = load_a_frag(abase, HID, k0, lane);
    v16bf b = load_b_frag(wbase, HID, k0, lane);
    acc = wmma_bf16(a, b, acc);
  }

  int n  = bn * 16 + (lane & 15);
  int g  = n >> 10;          // gate 0..3 (i,f,g,o)
  int jj = n & (HID - 1);    // hidden unit
  int hi = (lane >> 4) & 1;
  float bias = b_i[n] + b_h[n];
  #pragma unroll
  for (int r = 0; r < 8; ++r) {
    int row = bm * 16 + r + hi * 8;
    xg4[(((size_t)row * HID) + jj) * 4 + g] = acc[r] + bias;
  }
}

// ---------------------------------------------------------------------------
// Persistent recurrence kernel: all 256 steps, device-wide barrier per step.
// ---------------------------------------------------------------------------
__global__ void lstm_persistent_kernel(const unsigned short* __restrict__ whbf, // 4096x1024 bf16 [n][k]
                                       const float* __restrict__ xg4,           // 16384x1024x4 fp32
                                       const float* __restrict__ c0,            // 64x1024 fp32
                                       float* __restrict__ hseq,                // 64x256x1024 fp32
                                       unsigned short* __restrict__ hbf0,       // ping
                                       unsigned short* __restrict__ hbf1,       // pong
                                       float* __restrict__ h_final,
                                       float* __restrict__ c_final,
                                       unsigned int* __restrict__ bar) {
  __shared__ unsigned short sh_h[16 * HID];           // 32KB: this block's A tile

  int lane = threadIdx.x & 31;
  int wave = threadIdx.x >> 5;
  int tile = blockIdx.x * (NTHR / 32) + wave;         // 0..255
  int bm = tile >> 6;                                 // 0..3  batch tile (shared by block)
  int bj = tile & 63;                                 // 0..63 hidden tile

  const unsigned short* wb0 = whbf + (size_t)(0 * HID + bj * 16) * HID;
  const unsigned short* wb1 = whbf + (size_t)(1 * HID + bj * 16) * HID;
  const unsigned short* wb2 = whbf + (size_t)(2 * HID + bj * 16) * HID;
  const unsigned short* wb3 = whbf + (size_t)(3 * HID + bj * 16) * HID;

  int hi = (lane >> 4) & 1;
  int j  = bj * 16 + (lane & 15);

  // c state lives in registers for the entire sequence (one lane owns (b,j))
  float creg[8];
  #pragma unroll
  for (int r = 0; r < 8; ++r) {
    int b = bm * 16 + r + hi * 8;
    creg[r] = c0[(size_t)b * HID + j];
  }

  for (int t = 0; t < SEQ; ++t) {
    const unsigned short* hprev = (t & 1) ? hbf1 : hbf0;
    unsigned short*       hnext = (t & 1) ? hbf0 : hbf1;

    // ---- stage this block's 16-row h tile into LDS (32KB, 128B/thread) ----
#ifdef ASYNC_LDS_COPY
    {
      const char* src = (const char*)(hprev + (size_t)(bm * 16) * HID);
      #pragma unroll
      for (int i = 0; i < 8; ++i) {
        int off = (threadIdx.x + i * NTHR) * 16;
        // as(1)/as(3) int4 pointers built via integer casts (always legal; LDS
        // generic addr low 32 bits == LDS byte offset per ISA flat addressing)
        as1_v4i* gp = (as1_v4i*)(unsigned long long)(size_t)(src + off);
        as3_v4i* lp = (as3_v4i*)(unsigned int)(size_t)((const char*)sh_h + off);
        __builtin_amdgcn_global_load_async_to_lds_b128(gp, lp, 0, 0);
      }
#if __has_builtin(__builtin_amdgcn_s_wait_asynccnt)
      __builtin_amdgcn_s_wait_asynccnt(0);
#else
      asm volatile("s_wait_asynccnt 0x0" ::: "memory");
#endif
    }
#else
    {
      const uint4* src = (const uint4*)(hprev + (size_t)(bm * 16) * HID);
      uint4* dst = (uint4*)sh_h;
      #pragma unroll
      for (int i = 0; i < 8; ++i)
        dst[threadIdx.x + i * NTHR] = src[threadIdx.x + i * NTHR];
    }
#endif
    __syncthreads();

    // ---- 4-gate WMMA K-loop: A from LDS (shared), B from L2-resident Wh ----
    v8f acc0 = {}, acc1 = {}, acc2 = {}, acc3 = {};
    for (int k0 = 0; k0 < HID; k0 += 32) {
      v16bf a  = load_a_frag(sh_h, HID, k0, lane);    // ds_load_b128 path
      v16bf b0 = load_b_frag(wb0, HID, k0, lane);
      v16bf b1 = load_b_frag(wb1, HID, k0, lane);
      v16bf b2 = load_b_frag(wb2, HID, k0, lane);
      v16bf b3 = load_b_frag(wb3, HID, k0, lane);
      acc0 = wmma_bf16(a, b0, acc0);
      acc1 = wmma_bf16(a, b1, acc1);
      acc2 = wmma_bf16(a, b2, acc2);
      acc3 = wmma_bf16(a, b3, acc3);
    }

    // ---- fused LSTM elementwise epilogue (one b128 gate-vector per row) ----
    int tn = (t + 1 < SEQ) ? t + 1 : t;               // next step to prefetch
    #pragma unroll
    for (int r = 0; r < 8; ++r) {
      int b = bm * 16 + r + hi * 8;                   // C layout: M = r + 8*hi
      const float4 gv = *(const float4*)(xg4 + (((size_t)(b * SEQ + t) * HID) + j) * 4);
      float pi = acc0[r] + gv.x;
      float pf = acc1[r] + gv.y;
      float pg = acc2[r] + gv.z;
      float po = acc3[r] + gv.w;

      float ig = sigmoidf_fast(pi);
      float fg = sigmoidf_fast(pf);
      float gg = tanhf(pg);
      float og = sigmoidf_fast(po);

      float cn = fg * creg[r] + ig * gg;
      float hn = og * tanhf(cn);
      creg[r] = cn;

      size_t idx = (size_t)b * HID + j;
      hseq[((size_t)b * SEQ + t) * HID + j] = hn;
      hnext[idx] = f32_to_bf16(hn);
      if (t == SEQ - 1) { h_final[idx] = hn; c_final[idx] = cn; }

      // pull next timestep's gate vector toward this WGP while we sit in the barrier
      __builtin_prefetch(xg4 + (((size_t)(b * SEQ + tn) * HID) + j) * 4, 0, 0);
    }

    // ---- device-wide barrier (release h writes -> all WGPs, then acquire) ----
    __threadfence();                                  // release: push stores to L2
    __syncthreads();
    if (threadIdx.x == 0) {
      __hip_atomic_fetch_add(bar, 1u, __ATOMIC_RELEASE, __HIP_MEMORY_SCOPE_AGENT);
      unsigned int target = (unsigned int)(t + 1) * NBLK;
      while (__hip_atomic_load(bar, __ATOMIC_RELAXED, __HIP_MEMORY_SCOPE_AGENT) < target)
        __builtin_amdgcn_s_sleep(2);
    }
    __syncthreads();
    __threadfence();                                  // acquire: invalidate stale WGP$
  }
}

// ---------------------------------------------------------------------------
extern "C" void kernel_launch(void* const* d_in, const int* in_sizes, int n_in,
                              void* d_out, int out_size, void* d_ws, size_t ws_size,
                              hipStream_t stream) {
  const float* inputs = (const float*)d_in[0]; // (64,256,1024)
  const float* h0     = (const float*)d_in[1]; // (1,64,1024)
  const float* c0     = (const float*)d_in[2]; // (1,64,1024)
  const float* w_i    = (const float*)d_in[3]; // (4096,1024)
  const float* w_h    = (const float*)d_in[4]; // (4096,1024)
  const float* b_i    = (const float*)d_in[5]; // (4096,)
  const float* b_h    = (const float*)d_in[6]; // (4096,)

  float* out   = (float*)d_out;
  float* hseq  = out;                               // 64*256*1024
  float* h_fin = out + (size_t)BATCH * SEQ * HID;   // +16777216
  float* c_fin = h_fin + (size_t)BATCH * HID;       // +65536

  // workspace carve-up (bytes)
  char* ws = (char*)d_ws;
  unsigned short* xbf  = (unsigned short*)(ws);              //  32 MB: X bf16
  unsigned short* wibf = (unsigned short*)(ws + 33554432);   //   8 MB: Wi bf16
  unsigned short* whbf = (unsigned short*)(ws + 41943040);   //   8 MB: Wh bf16
  unsigned short* hbf0 = (unsigned short*)(ws + 50331648);   // 128 KB: h ping
  unsigned short* hbf1 = (unsigned short*)(ws + 50462720);   // 128 KB: h pong
  unsigned int*   bar  = (unsigned int*)  (ws + 50593792);   // barrier counter
  float*          xg4  = (float*)         (ws + 50855936);   // 256 MB: Xg fp32 (gate-interleaved)
  (void)ws_size; (void)in_sizes; (void)n_in; (void)out_size;

  // 1) precision conversion + state/barrier init
  cvt_bf16_kernel<<<4096, 256, 0, stream>>>(inputs, xbf, BATCH * SEQ * HID);
  cvt_bf16_kernel<<<2048, 256, 0, stream>>>(w_i, wibf, G4 * HID);
  cvt_bf16_kernel<<<2048, 256, 0, stream>>>(w_h, whbf, G4 * HID);
  init_state_kernel<<<256, 256, 0, stream>>>(h0, hbf0, bar, BATCH * HID);

  // 2) parallel input GEMM: Xg = X @ Wi^T + bi + bh  (gate-interleaved output)
  input_gemm_kernel<<<32768, 256, 0, stream>>>(xbf, wibf, b_i, b_h, xg4);

  // 3) single persistent kernel runs the whole 256-step recurrence
  lstm_persistent_kernel<<<NBLK, NTHR, 0, stream>>>(whbf, xg4, c0, hseq,
                                                    hbf0, hbf1, h_fin, c_fin, bar);
}
